// naiveLSTMCell_23974507446845
// MI455X (gfx1250) — matrix operand order repaired and checked by reading, not compiled, not run on hardware
//
#include <hip/hip_runtime.h>
#include <math.h>

// Problem constants (reference): B=16384, INP=HID=K=1024, 4 gates.
#define B_TOT   16384
#define K_DIM   1024
#define H_DIM   1024
#define NGATE   4

// ---------------- bf16 fast-path tiling ----------------
#define BM      128         // batch rows per workgroup
#define BH      64          // h columns per workgroup (per gate)
#define KT      32          // K per WMMA step (bf16 16x16x32)
#define NTHREADS 512        // 16 wave32s: 4 gates x 4 batch subtiles
#define KSTEPS  (2 * (K_DIM / KT))   // phase 0: inp*W_i, phase 1: h*W_h

typedef __bf16 bf16_t;
typedef __attribute__((ext_vector_type(16))) __bf16 v16bf;
typedef __attribute__((ext_vector_type(8)))  __bf16 v8bf;
typedef __attribute__((ext_vector_type(8)))  float  v8f;
typedef __attribute__((ext_vector_type(4)))  float  v4f;

union Frag16 { v16bf v; v8bf h[2]; };

__device__ __forceinline__ float fast_sigmoid(float x) {
    return 1.0f / (1.0f + __expf(-x));
}

// Low 32 bits of a generic pointer to LDS == workgroup-relative LDS byte
// address (ISA 10.2: LDS aperture keeps the LDS offset in addr[31:0]).
__device__ __forceinline__ unsigned lds_addr32(const void* p) {
    return (unsigned)(unsigned long long)p;
}

// ---------------------------------------------------------------------------
// Pre-pass: fp32 -> bf16 (vectorized, 8 elements/thread)
// ---------------------------------------------------------------------------
__global__ __launch_bounds__(256)
void cvt_f32_to_bf16(const float* __restrict__ src, bf16_t* __restrict__ dst,
                     int n8) {
    int i = blockIdx.x * 256 + threadIdx.x;
    if (i >= n8) return;
    size_t base = (size_t)i * 8;
    v4f a = *(const v4f*)&src[base];
    v4f b = *(const v4f*)&src[base + 4];
    v8bf o;
    #pragma unroll
    for (int j = 0; j < 4; ++j) { o[j] = (bf16_t)a[j]; o[4 + j] = (bf16_t)b[j]; }
    *(v8bf*)&dst[base] = o;
}

// ---------------------------------------------------------------------------
// Main fused kernel, bf16 operands. Staging uses CDNA5 async direct
// global->LDS DMA (GLOBAL_LOAD_ASYNC_TO_LDS_B128, ASYNCcnt-tracked):
// no VGPR staging, no ds_store in the loop.
// ---------------------------------------------------------------------------
__global__ __launch_bounds__(NTHREADS, 2)
void lstm_gemm_bf16(const bf16_t* __restrict__ Ax,   // inp   [B,K] bf16
                    const bf16_t* __restrict__ Ah,   // cur_h [B,K] bf16
                    const bf16_t* __restrict__ Wx,   // W_i   [4,H,K] bf16
                    const bf16_t* __restrict__ Wh,   // W_h   [4,H,K] bf16
                    const float*  __restrict__ b_i,  // [4,H]
                    const float*  __restrict__ b_h,  // [4,H]
                    const float*  __restrict__ cur_cell,
                    float* __restrict__ out_cell,
                    float* __restrict__ out_h)
{
    // Chunk = v8bf = 16 bytes. A: 512 chunks/buf, B: 1024 chunks/buf.
    __shared__ union {
        struct {
            v8bf A[2][BM * KT / 8];
            v8bf B[2][NGATE * BH * KT / 8];
        } stage;
        float pre[NGATE * BM * BH];          // 128 KB
    } sm;

    const int tid  = threadIdx.x;
    const int wid  = tid >> 5;
    const int lane = tid & 31;
    const int gate = wid >> 2;      // 0..3
    const int msub = wid & 3;       // batch subtile: rows msub*32 .. +31

    const int b0 = blockIdx.x * BM;
    const int h0 = blockIdx.y * BH;

    // ---- loop-invariant per-lane BYTE offsets for global loads ----
    const int arow = tid >> 2, aseg = tid & 3;
    const unsigned aByte = ((unsigned)(b0 + arow) * K_DIM + aseg * 8) * 2u;
    unsigned bByte[2];
    #pragma unroll
    for (int i = 0; i < 2; ++i) {
        int idx = i * NTHREADS + tid;       // 0..1023
        int row = idx >> 2;                 // 0..255 = g*64 + nn
        int seg = idx & 3;
        int g = row >> 6, nn = row & 63;
        bByte[i] = ((unsigned)(g * H_DIM + h0 + nn) * K_DIM + seg * 8) * 2u;
    }

    // ---- loop-invariant LDS destination addresses (per buffer) ----
    unsigned ldsA[2], ldsB0[2], ldsB1[2];
    #pragma unroll
    for (int bu = 0; bu < 2; ++bu) {
        ldsA[bu]  = lds_addr32(&sm.stage.A[bu][tid]);
        ldsB0[bu] = lds_addr32(&sm.stage.B[bu][tid]);
        ldsB1[bu] = lds_addr32(&sm.stage.B[bu][NTHREADS + tid]);
    }

    v8f acc[2][4];
    #pragma unroll
    for (int m = 0; m < 2; ++m)
        #pragma unroll
        for (int n = 0; n < 4; ++n)
            acc[m][n] = (v8f){0.f,0.f,0.f,0.f,0.f,0.f,0.f,0.f};

    // Issue the three async global->LDS B128 transfers for step s into buf.
    auto loadAsync = [&](int s, int buf) {
        const int phase = s >> 5;
        const unsigned kByte = (unsigned)(s & 31) * (KT * 2);  // uniform, SALU
        const char* __restrict__ pA = (const char*)(phase ? Ah : Ax) + kByte;
        const char* __restrict__ pW = (const char*)(phase ? Wh : Wx) + kByte;
        unsigned long long ga  = (unsigned long long)(pA + aByte);
        unsigned long long gb0 = (unsigned long long)(pW + bByte[0]);
        unsigned long long gb1 = (unsigned long long)(pW + bByte[1]);
        asm volatile("global_load_async_to_lds_b128 %0, %1, off"
                     :: "v"(ldsA[buf]),  "v"(ga)  : "memory");
        asm volatile("global_load_async_to_lds_b128 %0, %1, off"
                     :: "v"(ldsB0[buf]), "v"(gb0) : "memory");
        asm volatile("global_load_async_to_lds_b128 %0, %1, off"
                     :: "v"(ldsB1[buf]), "v"(gb1) : "memory");
    };
    auto waitAsync = [&]() {
        asm volatile("s_wait_asynccnt 0x0" ::: "memory");
    };

    // Fragment chunk indices (loop-invariant).
    const int a_kh = (lane < 16) ? 0 : 2;   // A elem offsets {0,16} + {0,8}
    const int b_kh = (lane < 16) ? 0 : 2;   // B elem offsets {0,16} + {0,8}

    auto computeStep = [&](int buf) {
        Frag16 a[2], b[4];
        #pragma unroll
        for (int m = 0; m < 2; ++m) {
            int c = (msub * 32 + m * 16 + (lane & 15)) * 4 + a_kh;
            a[m].h[0] = sm.stage.A[buf][c];
            a[m].h[1] = sm.stage.A[buf][c + 1];
        }
        #pragma unroll
        for (int n = 0; n < 4; ++n) {
            int c = (gate * BH + n * 16 + (lane & 15)) * 4 + b_kh;
            b[n].h[0] = sm.stage.B[buf][c];
            b[n].h[1] = sm.stage.B[buf][c + 1];
        }
        #pragma unroll
        for (int m = 0; m < 2; ++m)
            #pragma unroll
            for (int n = 0; n < 4; ++n)
                acc[m][n] = __builtin_amdgcn_wmma_f32_16x16x32_bf16(
                    false, a[m].v, false, b[n].v,
                    (short)0, acc[m][n], false, false);
    };

    // ---- async double-buffered main loop ----
    loadAsync(0, 0);
    waitAsync();
    __syncthreads();
    #pragma unroll 1
    for (int s = 0; s < KSTEPS - 1; ++s) {
        loadAsync(s + 1, (s + 1) & 1);   // DMA next tile while computing
        computeStep(s & 1);
        waitAsync();                     // own async loads complete
        __syncthreads();                 // all waves' tiles visible
    }
    computeStep((KSTEPS - 1) & 1);
    __syncthreads();   // stage buffers dead; union flips to `pre`

    // ---- accumulators -> LDS (gate-major) ----
    // C/D layout: VGPR j of lane l -> M = j + (l>=16 ? 8 : 0), N = l & 15.
    #pragma unroll
    for (int m = 0; m < 2; ++m)
        #pragma unroll
        for (int n = 0; n < 4; ++n)
            #pragma unroll
            for (int j = 0; j < 8; ++j) {
                int r = msub * 32 + m * 16 + j + ((lane >= 16) ? 8 : 0);
                int c = n * 16 + (lane & 15);
                sm.pre[(gate * BM + r) * BH + c] = acc[m][n][j];
            }
    __syncthreads();

    // ---- fused activation epilogue (consecutive lanes -> consecutive h) ----
    #pragma unroll
    for (int i = 0; i < (BM * BH) / NTHREADS; ++i) {
        int pos = i * NTHREADS + tid;
        int r = pos >> 6;
        int c = pos & (BH - 1);
        int h = h0 + c;
        size_t gidx = (size_t)(b0 + r) * H_DIM + h;

        float pi = sm.pre[(0 * BM + r) * BH + c] + b_i[0 * H_DIM + h] + b_h[0 * H_DIM + h];
        float pf = sm.pre[(1 * BM + r) * BH + c] + b_i[1 * H_DIM + h] + b_h[1 * H_DIM + h];
        float pg = sm.pre[(2 * BM + r) * BH + c] + b_i[2 * H_DIM + h] + b_h[2 * H_DIM + h];
        float po = sm.pre[(3 * BM + r) * BH + c] + b_i[3 * H_DIM + h] + b_h[3 * H_DIM + h];

        float ig = fast_sigmoid(pi);
        float fg = fast_sigmoid(pf);
        float gg = tanhf(pg);
        float og = fast_sigmoid(po);

        float nc = fg * cur_cell[gidx] + ig * gg;
        float nh = og * tanhf(nc);
        out_cell[gidx] = nc;
        out_h[gidx]    = nh;
    }
}

// ---------------------------------------------------------------------------
// Fallback (ws too small): fp32 loads + inline cvt, synchronous staging.
// ---------------------------------------------------------------------------
#define FBM 64
#define FNT 256

__global__ __launch_bounds__(FNT, 2)
void lstm_cell_fused_f32(const float* __restrict__ inp,
                         const float* __restrict__ cur_h,
                         const float* __restrict__ cur_cell,
                         const float* __restrict__ W_i,
                         const float* __restrict__ W_h,
                         const float* __restrict__ b_i,
                         const float* __restrict__ b_h,
                         float* __restrict__ out_cell,
                         float* __restrict__ out_h)
{
    __shared__ v8bf lsA[2][FBM * KT / 8];           // 256 chunks/buf
    __shared__ v8bf lsB[2][NGATE * BH * KT / 8];    // 1024 chunks/buf
    __shared__ float preLds[NGATE * FBM * BH];

    const int tid = threadIdx.x, wid = tid >> 5, lane = tid & 31;
    const int gate = wid >> 1, mh = wid & 1;
    const int b0 = blockIdx.x * FBM, h0 = blockIdx.y * BH;

    v8f acc[2][4];
    #pragma unroll
    for (int m = 0; m < 2; ++m)
        #pragma unroll
        for (int n = 0; n < 4; ++n)
            acc[m][n] = (v8f){0.f,0.f,0.f,0.f,0.f,0.f,0.f,0.f};

    v8bf ra, rb[4];

    auto loadStep = [&](int s) {
        const int phase = s >> 5;
        const unsigned kByte = (unsigned)(s & 31) * (KT * 4);   // f32 bytes
        const char* __restrict__ pA = (const char*)(phase ? cur_h : inp) + kByte;
        const char* __restrict__ pW = (const char*)(phase ? W_h : W_i) + kByte;
        {
            int row = tid >> 2, seg2 = tid & 3;
            unsigned off = ((unsigned)(b0 + row) * K_DIM + seg2 * 8) * 4u;
            v4f x = *(const v4f*)(pA + off);
            v4f y = *(const v4f*)(pA + off + 16);
            #pragma unroll
            for (int j = 0; j < 4; ++j) { ra[j] = (bf16_t)x[j]; ra[4+j] = (bf16_t)y[j]; }
        }
        #pragma unroll
        for (int i = 0; i < 4; ++i) {
            int idx = i * FNT + tid, row = idx >> 2, seg2 = idx & 3;
            int g = row >> 6, nn = row & 63;
            unsigned off = ((unsigned)(g * H_DIM + h0 + nn) * K_DIM + seg2 * 8) * 4u;
            v4f x = *(const v4f*)(pW + off);
            v4f y = *(const v4f*)(pW + off + 16);
            #pragma unroll
            for (int j = 0; j < 4; ++j) { rb[i][j] = (bf16_t)x[j]; rb[i][4+j] = (bf16_t)y[j]; }
        }
    };
    auto storeStep = [&](int buf) {
        lsA[buf][tid] = ra;
        #pragma unroll
        for (int i = 0; i < 4; ++i) lsB[buf][i * FNT + tid] = rb[i];
    };
    const int a_kh = (lane < 16) ? 0 : 2;
    const int b_kh = (lane < 16) ? 0 : 2;
    auto computeStep = [&](int buf) {
        Frag16 a[2], b[4];
        #pragma unroll
        for (int m = 0; m < 2; ++m) {
            int c = (mh * 32 + m * 16 + (lane & 15)) * 4 + a_kh;
            a[m].h[0] = lsA[buf][c];
            a[m].h[1] = lsA[buf][c + 1];
        }
        #pragma unroll
        for (int n = 0; n < 4; ++n) {
            int c = (gate * BH + n * 16 + (lane & 15)) * 4 + b_kh;
            b[n].h[0] = lsB[buf][c];
            b[n].h[1] = lsB[buf][c + 1];
        }
        #pragma unroll
        for (int m = 0; m < 2; ++m)
            #pragma unroll
            for (int n = 0; n < 4; ++n)
                acc[m][n] = __builtin_amdgcn_wmma_f32_16x16x32_bf16(
                    false, a[m].v, false, b[n].v, (short)0, acc[m][n], false, false);
    };

    loadStep(0); storeStep(0); __syncthreads();
    #pragma unroll 1
    for (int s = 0; s < KSTEPS - 1; ++s) {
        loadStep(s + 1); computeStep(s & 1); storeStep((s + 1) & 1);
        __syncthreads();
    }
    computeStep((KSTEPS - 1) & 1);

    #pragma unroll
    for (int m = 0; m < 2; ++m)
        #pragma unroll
        for (int n = 0; n < 4; ++n)
            #pragma unroll
            for (int j = 0; j < 8; ++j) {
                int r = mh * 32 + m * 16 + j + ((lane >= 16) ? 8 : 0);
                int c = n * 16 + (lane & 15);
                preLds[(gate * FBM + r) * BH + c] = acc[m][n][j];
            }
    __syncthreads();

    #pragma unroll
    for (int i = 0; i < (FBM * BH) / FNT; ++i) {
        int pos = i * FNT + tid;
        int r = pos >> 6, c = pos & (BH - 1), h = h0 + c;
        size_t gidx = (size_t)(b0 + r) * H_DIM + h;
        float pi = preLds[(0 * FBM + r) * BH + c] + b_i[h]             + b_h[h];
        float pf = preLds[(1 * FBM + r) * BH + c] + b_i[H_DIM + h]     + b_h[H_DIM + h];
        float pg = preLds[(2 * FBM + r) * BH + c] + b_i[2 * H_DIM + h] + b_h[2 * H_DIM + h];
        float po = preLds[(3 * FBM + r) * BH + c] + b_i[3 * H_DIM + h] + b_h[3 * H_DIM + h];
        float ig = fast_sigmoid(pi), fg = fast_sigmoid(pf);
        float gg = tanhf(pg),        og = fast_sigmoid(po);
        float nc = fg * cur_cell[gidx] + ig * gg;
        float nh = og * tanhf(nc);
        out_cell[gidx] = nc;
        out_h[gidx]    = nh;
    }
}

// ---------------------------------------------------------------------------
extern "C" void kernel_launch(void* const* d_in, const int* in_sizes, int n_in,
                              void* d_out, int out_size, void* d_ws, size_t ws_size,
                              hipStream_t stream) {
    const float* inp      = (const float*)d_in[0];
    const float* cur_h    = (const float*)d_in[1];
    const float* cur_cell = (const float*)d_in[2];
    const float* W_i      = (const float*)d_in[3];
    const float* W_h      = (const float*)d_in[4];
    const float* b_i      = (const float*)d_in[5];
    const float* b_h      = (const float*)d_in[6];

    float* out_cell = (float*)d_out;
    float* out_h    = out_cell + (size_t)B_TOT * H_DIM;

    const size_t nA = (size_t)B_TOT * K_DIM;          // 16M elements
    const size_t nW = (size_t)NGATE * H_DIM * K_DIM;  // 4M elements
    const size_t need = (2 * nA + 2 * nW) * sizeof(bf16_t);  // 80 MB

    if (ws_size >= need) {
        bf16_t* Ax = (bf16_t*)d_ws;
        bf16_t* Ah = Ax + nA;
        bf16_t* Wx = Ah + nA;
        bf16_t* Wh = Wx + nW;
        const int n8A = (int)(nA / 8), n8W = (int)(nW / 8);
        cvt_f32_to_bf16<<<(n8A + 255) / 256, 256, 0, stream>>>(inp,   Ax, n8A);
        cvt_f32_to_bf16<<<(n8A + 255) / 256, 256, 0, stream>>>(cur_h, Ah, n8A);
        cvt_f32_to_bf16<<<(n8W + 255) / 256, 256, 0, stream>>>(W_i,   Wx, n8W);
        cvt_f32_to_bf16<<<(n8W + 255) / 256, 256, 0, stream>>>(W_h,   Wh, n8W);

        dim3 grid(B_TOT / BM, H_DIM / BH);   // 128 x 16
        lstm_gemm_bf16<<<grid, NTHREADS, 0, stream>>>(
            Ax, Ah, Wx, Wh, b_i, b_h, cur_cell, out_cell, out_h);
    } else {
        dim3 grid(B_TOT / FBM, H_DIM / BH);  // 256 x 16
        lstm_cell_fused_f32<<<grid, FNT, 0, stream>>>(
            inp, cur_h, cur_cell, W_i, W_h, b_i, b_h, out_cell, out_h);
    }
}
